// BiasedAxialAttention_59974923321424
// MI455X (gfx1250) — matrix operand order repaired and checked by reading, not compiled, not run on hardware
//
#include <hip/hip_runtime.h>
#include <hip/hip_bf16.h>
#include <math.h>

#define LSEQ 384
#define DP   128
#define NH   4
#define DHD  32
#define KQ   (LSEQ*DHD)   // 12288

typedef __attribute__((ext_vector_type(16))) __bf16 v16bf;
typedef __attribute__((ext_vector_type(8)))  __bf16 v8bf;
typedef __attribute__((ext_vector_type(8)))  float  v8f;

__device__ __forceinline__ v8f wmma_bf16(v16bf a, v16bf b, v8f c) {
  return __builtin_amdgcn_wmma_f32_16x16x32_bf16(false, a, false, b, (short)0, c, false, false);
}

// ---- fragment loaders (contiguous k order shared by A and B, wave32) ----
// A 16x32 bf16: lane l holds row M = l&15; element e -> K = (l>=16?16:0) + e.
__device__ __forceinline__ v16bf load_a_frag(const __bf16* row_kb, int lane) {
  int hi = (lane >> 4) & 1;
  v8bf c0 = *(const v8bf*)(row_kb + hi * 16);
  v8bf c1 = *(const v8bf*)(row_kb + hi * 16 + 8);
  v16bf r;
#pragma unroll
  for (int e = 0; e < 8; ++e) { r[e] = c0[e]; r[e + 8] = c1[e]; }
  return r;
}

// B 32x16 bf16 from K-major storage (B^T[N][K]): one contiguous 32B chunk per lane.
__device__ __forceinline__ v16bf load_bt_frag(const __bf16* col_kb, int lane) {
  int hi = (lane >> 4) & 1;
  return *(const v16bf*)(col_kb + hi * 16);
}

// Pre-swizzled weight fragment: one aligned 32B vector load per lane.
__device__ __forceinline__ v16bf load_w_frag(const __bf16* wsw, int frag, int lane) {
  return *(const v16bf*)(wsw + ((size_t)frag * 32 + lane) * 16);
}

__device__ __forceinline__ float fast_sigmoid(float x) {
  return __builtin_amdgcn_rcpf(1.0f + __expf(-x));
}

// ============ Kernel 0: pack weights into WMMA B-fragment order (bf16) ============
__global__ __launch_bounds__(256) void k0_pack_weights(
    const float* __restrict__ Wq, const float* __restrict__ Wk,
    const float* __restrict__ Wv, const float* __restrict__ Wg,
    const float* __restrict__ Wo,
    __bf16* __restrict__ w4sw, __bf16* __restrict__ wosw) {
  int id = blockIdx.x * blockDim.x + threadIdx.x;   // 0..5119
  int l = id & 31;
  int frag = id >> 5;                               // 0..159
  int hi16 = ((l >> 4) & 1) * 16;
  if (frag < 128) {
    int nt = frag >> 2, kb = frag & 3;
    int quad = nt >> 3;
    const float* Wm = (quad == 0) ? Wq : (quad == 1) ? Wk : (quad == 2) ? Wv : Wg;
    int col = (nt & 7) * 16 + (l & 15);
    int k0  = kb * 32 + hi16;
    __bf16* dst = w4sw + ((size_t)frag * 32 + l) * 16;
#pragma unroll
    for (int e = 0; e < 16; ++e) dst[e] = (__bf16)Wm[(size_t)(k0 + e) * DP + col];
  } else {
    int f2 = frag - 128;
    int nt = f2 >> 2, kb = f2 & 3;
    int col = nt * 16 + (l & 15);
    int k0  = kb * 32 + hi16;
    __bf16* dst = wosw + ((size_t)f2 * 32 + l) * 16;
#pragma unroll
    for (int e = 0; e < 16; ++e) dst[e] = (__bf16)Wo[(size_t)(k0 + e) * DP + col];
  }
}

// ================= Kernel 1: LayerNorm + Q/K/V/G projections =================
__global__ __launch_bounds__(256) void k1_ln_proj(
    const float* __restrict__ pair, const float* __restrict__ ln_g, const float* __restrict__ ln_b,
    const __bf16* __restrict__ w4sw, const float* __restrict__ bg,
    __bf16* __restrict__ qws, __bf16* __restrict__ kws,
    __bf16* __restrict__ vtws, __bf16* __restrict__ gws) {
  __shared__ __bf16 sx[32][DP];

  int tid = threadIdx.x;
  int t0  = blockIdx.x * 32;
  int r   = tid >> 3;
  int fo  = (tid & 7) * 16;
  int t   = t0 + r;
  int n   = t / LSEQ, i = t % LSEQ;

  const float4* s4 = (const float4*)(pair + ((size_t)i * LSEQ + n) * DP + fo);
  float v[16];
  float s = 0.f, sq = 0.f;
#pragma unroll
  for (int q = 0; q < 4; ++q) {
    float4 x = s4[q];
    v[q*4+0] = x.x; v[q*4+1] = x.y; v[q*4+2] = x.z; v[q*4+3] = x.w;
    s += x.x + x.y + x.z + x.w;
    sq += x.x*x.x + x.y*x.y + x.z*x.z + x.w*x.w;
  }
  // 8-lane in-wave reduction (row group is wave-contiguous)
#pragma unroll
  for (int m = 1; m < 8; m <<= 1) {
    s  += __shfl_xor(s,  m, 32);
    sq += __shfl_xor(sq, m, 32);
  }
  float mean = s * (1.0f / DP);
  float var  = sq * (1.0f / DP) - mean * mean;
  float rstd = rsqrtf(var + 1e-5f);
#pragma unroll
  for (int e = 0; e < 16; ++e) {
    float xh = (v[e] - mean) * rstd * ln_g[fo + e] + ln_b[fo + e];
    sx[r][fo + e] = (__bf16)xh;
  }
  __syncthreads();

  // GEMM: [32 x 128] x [128 x 512]; wave w handles quadrant ng = w>>1 (uniform)
  int w = tid >> 5, l = tid & 31;
  int mt = w & 1;
  int ng = w >> 1;                 // 0:Q 1:K 2:V 3:G  (wave-uniform)
  v16bf afr[4];
#pragma unroll
  for (int kb = 0; kb < 4; ++kb)
    afr[kb] = load_a_frag(&sx[mt * 16 + (l & 15)][kb * 32], l);

  const float qscale = rsqrtf(32.0f * 384.0f);
  int hi8 = ((l >> 4) & 1) << 3;

  auto gemm8 = [&](auto&& store) {
#pragma unroll
    for (int ntl = 0; ntl < 8; ++ntl) {
      int nt = ng * 8 + ntl;
      v8f acc = {};
#pragma unroll
      for (int kb = 0; kb < 4; ++kb)
        acc = wmma_bf16(afr[kb], load_w_frag(w4sw, nt * 4 + kb, l), acc);
      int col = ntl * 16 + (l & 15);          // 0..127 within quadrant
#pragma unroll
      for (int rr = 0; rr < 8; ++rr) {
        int tt = t0 + mt * 16 + rr + hi8;
        store(tt, col, acc[rr]);
      }
    }
  };

  if (ng == 0) {
    gemm8([&](int tt, int col, float val) {
      int nn = tt / LSEQ, ii = tt % LSEQ;
      qws[((size_t)(col >> 5) * LSEQ + ii) * KQ + nn * DHD + (col & 31)] = (__bf16)(val * qscale);
    });
  } else if (ng == 1) {
    gemm8([&](int tt, int col, float val) {
      int nn = tt / LSEQ, ii = tt % LSEQ;
      kws[((size_t)(col >> 5) * LSEQ + ii) * KQ + nn * DHD + (col & 31)] = (__bf16)val;
    });
  } else if (ng == 2) {
    gemm8([&](int tt, int col, float val) {
      int nn = tt / LSEQ, ii = tt % LSEQ;
      vtws[((size_t)(col >> 5) * KQ + nn * DHD + (col & 31)) * LSEQ + ii] = (__bf16)val;
    });
  } else {
    gemm8([&](int tt, int col, float val) {
      gws[(size_t)tt * DP + col] = (__bf16)fast_sigmoid(val + bg[col]);
    });
  }
}

// ================= Kernel 2: logits[h][i][j] = sum_{K=12288} Q[i,K]*K[j,K] =================
__global__ __launch_bounds__(256) void k2_logits(
    const __bf16* __restrict__ qws, const __bf16* __restrict__ kws, float* __restrict__ logits) {
  int h   = blockIdx.z;
  int ib0 = blockIdx.y * 64;
  int jb0 = blockIdx.x * 64;
  int tid = threadIdx.x, w = tid >> 5, l = tid & 31;
  int wm = w & 3;
  int wn = w >> 2;

  const __bf16* aRow  = qws + (size_t)h * LSEQ * KQ + (size_t)(ib0 + wm * 16 + (l & 15)) * KQ;
  const __bf16* bCol0 = kws + (size_t)h * LSEQ * KQ + (size_t)(jb0 + wn * 32 + (l & 15)) * KQ;
  const __bf16* bCol1 = bCol0 + (size_t)16 * KQ;

  v8f acc0 = {}, acc1 = {};
  for (int kb = 0; kb < KQ; kb += 32) {
    __builtin_prefetch(aRow + kb + 512, 0, 0);
    v16bf a = load_a_frag(aRow + kb, l);
    acc0 = wmma_bf16(a, load_bt_frag(bCol0 + kb, l), acc0);
    acc1 = wmma_bf16(a, load_bt_frag(bCol1 + kb, l), acc1);
  }
  float* Cp = logits + (size_t)h * LSEQ * LSEQ;
  int hi8 = ((l >> 4) & 1) << 3;
  int jc = jb0 + wn * 32 + (l & 15);
#pragma unroll
  for (int rr = 0; rr < 8; ++rr) {
    int i = ib0 + wm * 16 + rr + hi8;
    Cp[(size_t)i * LSEQ + jc]      = acc0[rr];
    Cp[(size_t)i * LSEQ + jc + 16] = acc1[rr];
  }
}

// ================= Kernel 3: softmax over j with fused bias@Wb =================
__global__ __launch_bounds__(128) void k3_softmax(
    const float* __restrict__ logits, const float* __restrict__ bias, const float* __restrict__ Wb,
    __bf16* __restrict__ attn) {
  int h = blockIdx.y;
  int i = blockIdx.x;
  int tid = threadIdx.x;
  __shared__ float red[128];
  __shared__ float sWb[DP];
  sWb[tid] = Wb[tid * NH + h];          // blockDim == DP
  __syncthreads();

  float lv[3];
#pragma unroll
  for (int u = 0; u < 3; ++u) {
    int j = tid + u * 128;
    const float4* bp4 = (const float4*)(bias + ((size_t)i * LSEQ + j) * DP);
    const float4* wb4 = (const float4*)sWb;
    float bsum = 0.f;
#pragma unroll 8
    for (int d = 0; d < DP / 4; ++d) {
      float4 x = bp4[d], y = wb4[d];
      bsum += x.x * y.x + x.y * y.y + x.z * y.z + x.w * y.w;
    }
    lv[u] = logits[((size_t)h * LSEQ + i) * LSEQ + j] + bsum;
  }
  float m = fmaxf(fmaxf(lv[0], lv[1]), lv[2]);
  red[tid] = m; __syncthreads();
  for (int s = 64; s > 0; s >>= 1) { if (tid < s) red[tid] = fmaxf(red[tid], red[tid + s]); __syncthreads(); }
  m = red[0]; __syncthreads();
  float e[3], ssum = 0.f;
#pragma unroll
  for (int u = 0; u < 3; ++u) { e[u] = __expf(lv[u] - m); ssum += e[u]; }
  red[tid] = ssum; __syncthreads();
  for (int s = 64; s > 0; s >>= 1) { if (tid < s) red[tid] += red[tid + s]; __syncthreads(); }
  float inv = __builtin_amdgcn_rcpf(red[0]);
#pragma unroll
  for (int u = 0; u < 3; ++u) {
    int j = tid + u * 128;
    attn[((size_t)h * LSEQ + i) * LSEQ + j] = (__bf16)(e[u] * inv);
  }
}

// ================= Kernel 4: o[i,(k,d)] = sum_j attn[h][i][j] * vT[h][(k,d)][j] =================
__global__ __launch_bounds__(256) void k4_av(
    const __bf16* __restrict__ attn, const __bf16* __restrict__ vtws, __bf16* __restrict__ ows) {
  int h   = blockIdx.z;
  int ib0 = blockIdx.y * 64;
  int nb0 = blockIdx.x * 64;
  int tid = threadIdx.x, w = tid >> 5, l = tid & 31;
  int wm  = w & 3;
  int wn2 = w >> 2;
  int n0  = nb0 + wn2 * 32;

  const __bf16* aRow  = attn + (size_t)h * LSEQ * LSEQ + (size_t)(ib0 + wm * 16 + (l & 15)) * LSEQ;
  const __bf16* Bt    = vtws + (size_t)h * KQ * LSEQ;
  const __bf16* bCol0 = Bt + (size_t)(n0 + (l & 15)) * LSEQ;
  const __bf16* bCol1 = Bt + (size_t)(n0 + 16 + (l & 15)) * LSEQ;

  v8f acc0 = {}, acc1 = {};
#pragma unroll 4
  for (int kb = 0; kb < LSEQ; kb += 32) {
    v16bf a = load_a_frag(aRow + kb, l);
    acc0 = wmma_bf16(a, load_bt_frag(bCol0 + kb, l), acc0);
    acc1 = wmma_bf16(a, load_bt_frag(bCol1 + kb, l), acc1);
  }
  int hi8 = ((l >> 4) & 1) << 3;
#pragma unroll
  for (int rr = 0; rr < 8; ++rr) {
    int i = ib0 + wm * 16 + rr + hi8;
    int N0 = n0 + (l & 15);
    int N1 = N0 + 16;
    ows[(((size_t)i * LSEQ + (N0 >> 5)) * NH + h) * DHD + (N0 & 31)] = (__bf16)acc0[rr];
    ows[(((size_t)i * LSEQ + (N1 >> 5)) * NH + h) * DHD + (N1 & 31)] = (__bf16)acc1[rr];
  }
}

// ================= Kernel 5: out = (g*o) @ Wo + bo, transposed store =================
__global__ __launch_bounds__(256) void k5_out(
    const __bf16* __restrict__ gws, const __bf16* __restrict__ ows,
    const __bf16* __restrict__ wosw, const float* __restrict__ bo, float* __restrict__ out) {
  __shared__ __bf16 sm[32][DP];
  int tid = threadIdx.x;
  int t0  = blockIdx.x * 32;
  int r   = tid >> 3, fo = (tid & 7) * 16;
  size_t base = (size_t)(t0 + r) * DP + fo;
  v8bf g0 = *(const v8bf*)(gws + base), g1 = *(const v8bf*)(gws + base + 8);
  v8bf o0 = *(const v8bf*)(ows + base), o1 = *(const v8bf*)(ows + base + 8);
#pragma unroll
  for (int e = 0; e < 8; ++e) {
    sm[r][fo + e]     = (__bf16)((float)g0[e] * (float)o0[e]);
    sm[r][fo + 8 + e] = (__bf16)((float)g1[e] * (float)o1[e]);
  }
  __syncthreads();

  int w = tid >> 5, l = tid & 31;
  int nt = w;
  for (int mt = 0; mt < 2; ++mt) {
    v8f acc = {};
#pragma unroll
    for (int kb = 0; kb < 4; ++kb) {
      v16bf a = load_a_frag(&sm[mt * 16 + (l & 15)][kb * 32], l);
      acc = wmma_bf16(a, load_w_frag(wosw, nt * 4 + kb, l), acc);
    }
    int col = nt * 16 + (l & 15);
    float bc = bo[col];
    int hi8 = ((l >> 4) & 1) << 3;
#pragma unroll
    for (int rr = 0; rr < 8; ++rr) {
      int t = t0 + mt * 16 + rr + hi8;
      int a_ = t / LSEQ, c_ = t % LSEQ;
      out[((size_t)c_ * LSEQ + a_) * DP + col] = acc[rr] + bc;
    }
  }
}

extern "C" void kernel_launch(void* const* d_in, const int* in_sizes, int n_in,
                              void* d_out, int out_size, void* d_ws, size_t ws_size,
                              hipStream_t stream) {
  (void)in_sizes; (void)n_in; (void)out_size; (void)ws_size;
  const float* pair = (const float*)d_in[0];
  const float* bias = (const float*)d_in[1];
  const float* ln_g = (const float*)d_in[2];
  const float* ln_b = (const float*)d_in[3];
  const float* Wq   = (const float*)d_in[4];
  const float* Wk   = (const float*)d_in[5];
  const float* Wv   = (const float*)d_in[6];
  const float* Wb   = (const float*)d_in[7];
  const float* Wg   = (const float*)d_in[8];
  const float* bg   = (const float*)d_in[9];
  const float* Wo   = (const float*)d_in[10];
  const float* bo   = (const float*)d_in[11];
  float* out = (float*)d_out;

  char* ws = (char*)d_ws;
  const size_t NPOS = (size_t)LSEQ * LSEQ;        // 147456
  const size_t QSZ  = (size_t)NH * LSEQ * KQ;     // 18,874,368 elems
  __bf16* qws   = (__bf16*)ws;  ws += QSZ * 2;
  __bf16* kws   = (__bf16*)ws;  ws += QSZ * 2;
  __bf16* vtws  = (__bf16*)ws;  ws += QSZ * 2;
  __bf16* gws   = (__bf16*)ws;  ws += NPOS * DP * 2;
  __bf16* ows   = (__bf16*)ws;  ws += NPOS * DP * 2;
  __bf16* attn  = (__bf16*)ws;  ws += (size_t)NH * LSEQ * LSEQ * 2;
  __bf16* w4sw  = (__bf16*)ws;  ws += (size_t)128 * 32 * 16 * 2;   // 65536 elems
  __bf16* wosw  = (__bf16*)ws;  ws += (size_t)32 * 32 * 16 * 2;    // 16384 elems
  float*  logits = (float*)ws;  ws += (size_t)NH * LSEQ * LSEQ * 4;

  k0_pack_weights<<<20, 256, 0, stream>>>(Wq, Wk, Wv, Wg, Wo, w4sw, wosw);
  k1_ln_proj<<<(int)(NPOS / 32), 256, 0, stream>>>(pair, ln_g, ln_b, w4sw, bg,
                                                   qws, kws, vtws, gws);
  k2_logits<<<dim3(6, 6, NH), 256, 0, stream>>>(qws, kws, logits);
  k3_softmax<<<dim3(LSEQ, NH), 128, 0, stream>>>(logits, bias, Wb, attn);
  k4_av<<<dim3(192, 6, NH), 256, 0, stream>>>(attn, vtws, ows);
  k5_out<<<(int)(NPOS / 32), 256, 0, stream>>>(gws, ows, wosw, bo, out);
}
